// MultiHeadAttention_55697135895243
// MI455X (gfx1250) — matrix operand order repaired and checked by reading, not compiled, not run on hardware
//
#include <hip/hip_runtime.h>
#include <hip/hip_bf16.h>

typedef __bf16 bf16;
typedef __attribute__((ext_vector_type(4)))  __bf16 v4bf;
typedef __attribute__((ext_vector_type(8)))  __bf16 v8bf;
typedef __attribute__((ext_vector_type(16))) __bf16 v16bf;
typedef __attribute__((ext_vector_type(8)))  float  v8f;

union V16 { v16bf v; v8bf h[2]; };

__device__ __forceinline__ v8bf cvt8(const float4& a, const float4& b) {
  v8bf r;
  r[0] = (__bf16)a.x; r[1] = (__bf16)a.y; r[2] = (__bf16)a.z; r[3] = (__bf16)a.w;
  r[4] = (__bf16)b.x; r[5] = (__bf16)b.y; r[6] = (__bf16)b.z; r[7] = (__bf16)b.w;
  return r;
}

__device__ __forceinline__ v8f wmma_bf16(v16bf a, v16bf b, v8f c) {
  // D = A(16x32 bf16) * B(32x16 bf16) + C(16x16 f32)
  return __builtin_amdgcn_wmma_f32_16x16x32_bf16(
      /*neg_a=*/false, a, /*neg_b=*/false, b,
      /*c_mod=*/(short)0, c, /*reuse_a=*/false, /*reuse_b=*/false);
}

// LDS byte offset of a shared-memory pointer (low 32 bits of the flat addr).
__device__ __forceinline__ uint32_t lds_off(const void* p) {
  return (uint32_t)(uintptr_t)p;
}

// Async DMA: 16 bytes per lane, global -> LDS, tracked by ASYNCcnt.
__device__ __forceinline__ void async_copy_b128(uint32_t dst_lds, const void* src) {
  asm volatile("global_load_async_to_lds_b128 %0, %1, off"
               :: "v"(dst_lds), "v"(src) : "memory");
}

__device__ __forceinline__ void wait_async0() {
  asm volatile("s_wait_asynccnt 0x0" ::: "memory");
}

// ---------------------------------------------------------------------------
// Kernel 0: weight transpose + f32->bf16 convert: Wt[n][k] = (bf16)W[k][n].
// 32x32 tiles through LDS; coalesced read along n, coalesced write along k.
// ---------------------------------------------------------------------------
__global__ __launch_bounds__(256) void wt_convert_kernel(
    const float* __restrict__ W, bf16* __restrict__ Wt, int K, int Nn)
{
  __shared__ alignas(16) bf16 Ts[32][40];
  const int tid = threadIdx.x;
  const int k0 = blockIdx.y << 5;
  const int n0 = blockIdx.x << 5;
  {
    const int k = tid >> 3;            // 0..31
    const int n4 = (tid & 7) << 2;     // 0..28 step 4
    const float4 w = *(const float4*)(W + (size_t)(k0 + k) * Nn + n0 + n4);
    Ts[n4 + 0][k] = (__bf16)w.x;
    Ts[n4 + 1][k] = (__bf16)w.y;
    Ts[n4 + 2][k] = (__bf16)w.z;
    Ts[n4 + 3][k] = (__bf16)w.w;
  }
  __syncthreads();
  {
    const int n = tid >> 3;
    const int k4 = (tid & 7) << 2;
    *(v4bf*)(Wt + (size_t)(n0 + n) * K + k0 + k4) = *(const v4bf*)&Ts[n][k4];
  }
}

// ---------------------------------------------------------------------------
// Kernel 1: qkv = x @ W_attn + b_attn, scattered to head-major bf16 Q/K/V.
// X: [8192 x 1024] f32 (converted inline); Wt: [3072 x 1024] bf16 (pre-
// transposed) staged via async DMA. Double-buffered 128x128x32 pipeline.
// Q pre-scaled by 1/sqrt(Dh) = 0.125.
// ---------------------------------------------------------------------------
__global__ __launch_bounds__(256) void qkv_gemm_kernel(
    const float* __restrict__ X, const bf16* __restrict__ Wt,
    const float* __restrict__ bias,
    bf16* __restrict__ Qb, bf16* __restrict__ Kb, bf16* __restrict__ Vb)
{
  constexpr int LDT = 40;  // bf16 row stride, 80B (16B multiple)
  __shared__ alignas(16) bf16 As[2][128][LDT];   // [m][k]
  __shared__ alignas(16) bf16 Bs[2][128][LDT];   // [n][k]

  const int tid = threadIdx.x;
  const int wave = tid >> 5, lane = tid & 31;
  const int laneN = lane & 15, grp = lane >> 4;
  const int m0 = blockIdx.y * 128;
  const int n0 = blockIdx.x * 128;
  const int waveM = wave & 3;   // 4 waves x 32 rows
  const int waveN = wave >> 2;  // 2 waves x 64 cols

  const v8f vzero = {0.f, 0.f, 0.f, 0.f, 0.f, 0.f, 0.f, 0.f};
  v8f acc[2][4];
#pragma unroll
  for (int i = 0; i < 2; ++i)
#pragma unroll
    for (int j = 0; j < 4; ++j) acc[i][j] = vzero;

  const int arow = tid >> 1, acol = (tid & 1) << 4;  // A: 16 f32 / thread
  const int bn = tid >> 1,   bk = (tid & 1) << 4;    // B: 16 bf16 / thread

  const float* aptr = X + (size_t)(m0 + arow) * 1024 + acol;
  const bf16*  bptr = Wt + (size_t)(n0 + bn) * 1024 + bk;

  // Prologue: stage tile 0 (B via async DMA, A via convert path).
  async_copy_b128(lds_off(&Bs[0][bn][bk]),     bptr);
  async_copy_b128(lds_off(&Bs[0][bn][bk + 8]), bptr + 8);
  {
    float4 a0 = *(const float4*)(aptr + 0);
    float4 a1 = *(const float4*)(aptr + 4);
    float4 a2 = *(const float4*)(aptr + 8);
    float4 a3 = *(const float4*)(aptr + 12);
    *(v8bf*)&As[0][arow][acol + 0] = cvt8(a0, a1);
    *(v8bf*)&As[0][arow][acol + 8] = cvt8(a2, a3);
  }
  wait_async0();
  __syncthreads();

  int cur = 0;
  for (int k0 = 0; k0 < 1024; k0 += 32) {
    const int nxt = cur ^ 1;
    const bool has_next = (k0 + 32) < 1024;  // wave-uniform
    float4 a0{}, a1{}, a2{}, a3{};
    if (has_next) {  // kick off DMA + global loads for tile t+1
      const bf16* bp = bptr + (k0 + 32);
      async_copy_b128(lds_off(&Bs[nxt][bn][bk]),     bp);
      async_copy_b128(lds_off(&Bs[nxt][bn][bk + 8]), bp + 8);
      const float* ap = aptr + (k0 + 32);
      a0 = *(const float4*)(ap + 0);
      a1 = *(const float4*)(ap + 4);
      a2 = *(const float4*)(ap + 8);
      a3 = *(const float4*)(ap + 12);
    }

    // Compute on tile t.
    V16 af[2];
#pragma unroll
    for (int mt = 0; mt < 2; ++mt) {
      const int r = waveM * 32 + mt * 16 + laneN;
      af[mt].h[0] = *(const v8bf*)&As[cur][r][grp * 8];
      af[mt].h[1] = *(const v8bf*)&As[cur][r][16 + grp * 8];
    }
#pragma unroll
    for (int nt = 0; nt < 4; ++nt) {
      V16 bfr;
      const int c = waveN * 64 + nt * 16 + laneN;
      bfr.h[0] = *(const v8bf*)&Bs[cur][c][grp * 8];
      bfr.h[1] = *(const v8bf*)&Bs[cur][c][16 + grp * 8];
#pragma unroll
      for (int mt = 0; mt < 2; ++mt)
        acc[mt][nt] = wmma_bf16(af[mt].v, bfr.v, acc[mt][nt]);
    }

    if (has_next) {  // convert + stage A tile t+1
      *(v8bf*)&As[nxt][arow][acol + 0] = cvt8(a0, a1);
      *(v8bf*)&As[nxt][arow][acol + 8] = cvt8(a2, a3);
    }
    wait_async0();
    __syncthreads();
    cur = nxt;
  }

  // Epilogue: bias, convert, scatter to [B,H,T,Dh] bf16 buffers.
#pragma unroll
  for (int nt = 0; nt < 4; ++nt) {
    const int gnBase = n0 + waveN * 64 + nt * 16;  // 16-aligned -> uniform
    const int gn = gnBase + laneN;
    const float bv = bias[gn];
    const int which = gnBase >> 10;                // uniform: 0=q 1=k 2=v
    const int rem = gn & 1023;
    const int hh = rem >> 6;
    const int dd = rem & 63;
    bf16* dstbuf = (which == 0) ? Qb : (which == 1) ? Kb : Vb;
    const float scale = (which == 0) ? 0.125f : 1.0f;
#pragma unroll
    for (int mt = 0; mt < 2; ++mt) {
#pragma unroll
      for (int r = 0; r < 8; ++r) {
        const int gm = m0 + waveM * 32 + mt * 16 + r + (grp << 3);
        const int bI = gm >> 11;
        const int t = gm & 2047;
        const size_t dst = (((size_t)bI * 16 + hh) * 2048 + t) * 64 + dd;
        dstbuf[dst] = (__bf16)((acc[mt][nt][r] + bv) * scale);
      }
    }
  }
}

// ---------------------------------------------------------------------------
// Kernel 2: causal flash attention. Block = (b,h) x 128 Q rows; each wave
// owns 16 Q rows; KV blocks of 64 keys. All GEMMs via bf16 WMMA, f32 acc.
// ---------------------------------------------------------------------------
__global__ __launch_bounds__(256) void flash_attn_kernel(
    const bf16* __restrict__ Qb, const bf16* __restrict__ Kb,
    const bf16* __restrict__ Vb, bf16* __restrict__ Yb)
{
  constexpr int LDA = 72;  // 144B row stride (16B multiple)
  __shared__ alignas(16) bf16 Qs[128][LDA];     // [q][d]
  __shared__ alignas(16) bf16 Ks[64][LDA];      // [key][d]
  __shared__ alignas(16) bf16 Vs[64][LDA];      // [d][key] (transposed)
  __shared__ alignas(16) bf16 Ps[8][16][LDA];   // per-wave P: [q][key]

  const int tid = threadIdx.x;
  const int wave = tid >> 5, lane = tid & 31;
  const int laneN = lane & 15, grp = lane >> 4;
  const int bh = blockIdx.x;
  const int b = bh >> 4, h = bh & 15;
  const int q0blk = blockIdx.y * 128;
  const size_t head = (size_t)bh * 2048 * 64;

  { // stage Q block (128 x 64 bf16)
    const int row = tid >> 1;
    const int half = (tid & 1) << 5;
    const bf16* qp = Qb + head + (size_t)(q0blk + row) * 64 + half;
#pragma unroll
    for (int i = 0; i < 4; ++i)
      *(v8bf*)&Qs[row][half + i * 8] = *(const v8bf*)(qp + i * 8);
  }
  __syncthreads();

  V16 qa[2];
  {
    const int qrow = wave * 16 + laneN;
#pragma unroll
    for (int ks = 0; ks < 2; ++ks) {
      qa[ks].h[0] = *(const v8bf*)&Qs[qrow][ks * 32 + grp * 8];
      qa[ks].h[1] = *(const v8bf*)&Qs[qrow][ks * 32 + 16 + grp * 8];
    }
  }

  const v8f vzero = {0.f, 0.f, 0.f, 0.f, 0.f, 0.f, 0.f, 0.f};
  float m_s[8], l_s[8];
  v8f o[4];
#pragma unroll
  for (int r = 0; r < 8; ++r) { m_s[r] = -1e30f; l_s[r] = 0.0f; }
#pragma unroll
  for (int nt = 0; nt < 4; ++nt) o[nt] = vzero;

  const int q0w = q0blk + wave * 16;
  const int nkv = (q0blk >> 6) + 2;  // causal: blocks up to this Q tile

  for (int jb = 0; jb < nkv; ++jb) {
    const int kb = jb << 6;
    __syncthreads();
    { // stage K (row-major) and V (transposed)
      const int row = tid >> 2;
      const int qtr = (tid & 3) << 4;
      const bf16* kp = Kb + head + (size_t)(kb + row) * 64 + qtr;
      *(v8bf*)&Ks[row][qtr + 0] = *(const v8bf*)(kp + 0);
      *(v8bf*)&Ks[row][qtr + 8] = *(const v8bf*)(kp + 8);
      const bf16* vp = Vb + head + (size_t)(kb + row) * 64 + qtr;
      union { v8bf hv[2]; bf16 e[16]; } vv;
      vv.hv[0] = *(const v8bf*)(vp + 0);
      vv.hv[1] = *(const v8bf*)(vp + 8);
#pragma unroll
      for (int i = 0; i < 16; ++i) Vs[qtr + i][row] = vv.e[i];
    }
    __syncthreads();

    if (q0w + 15 >= kb) {  // wave-uniform causal skip (EXEC stays full)
      // S = Q K^T
      v8f s[4];
#pragma unroll
      for (int nt = 0; nt < 4; ++nt) {
        s[nt] = vzero;
#pragma unroll
        for (int ks = 0; ks < 2; ++ks) {
          V16 kf;
          const bf16* kr = &Ks[nt * 16 + laneN][ks * 32 + grp * 8];
          kf.h[0] = *(const v8bf*)(kr + 0);
          kf.h[1] = *(const v8bf*)(kr + 16);
          s[nt] = wmma_bf16(qa[ks].v, kf.v, s[nt]);
        }
      }
      // causal mask + running row max
      float mnew[8];
#pragma unroll
      for (int r = 0; r < 8; ++r) mnew[r] = m_s[r];
#pragma unroll
      for (int nt = 0; nt < 4; ++nt) {
        const int kg = kb + nt * 16 + laneN;
#pragma unroll
        for (int r = 0; r < 8; ++r) {
          const int qg = q0w + r + (grp << 3);
          const float val = (kg > qg) ? -1e30f : s[nt][r];
          s[nt][r] = val;
          mnew[r] = fmaxf(mnew[r], val);
        }
      }
#pragma unroll
      for (int msk = 1; msk < 16; msk <<= 1)
#pragma unroll
        for (int r = 0; r < 8; ++r)
          mnew[r] = fmaxf(mnew[r], __shfl_xor(mnew[r], msk, 32));

      // P = exp(S - m), row sums
      float lsum[8];
#pragma unroll
      for (int r = 0; r < 8; ++r) lsum[r] = 0.0f;
#pragma unroll
      for (int nt = 0; nt < 4; ++nt)
#pragma unroll
        for (int r = 0; r < 8; ++r) {
          const float p =
              __builtin_amdgcn_exp2f((s[nt][r] - mnew[r]) * 1.4426950408889634f);
          s[nt][r] = p;
          lsum[r] += p;
        }
#pragma unroll
      for (int msk = 1; msk < 16; msk <<= 1)
#pragma unroll
        for (int r = 0; r < 8; ++r)
          lsum[r] += __shfl_xor(lsum[r], msk, 32);

      // rescale running state
#pragma unroll
      for (int r = 0; r < 8; ++r) {
        const float alpha =
            __builtin_amdgcn_exp2f((m_s[r] - mnew[r]) * 1.4426950408889634f);
        m_s[r] = mnew[r];
        l_s[r] = l_s[r] * alpha + lsum[r];
#pragma unroll
        for (int nt = 0; nt < 4; ++nt) o[nt][r] *= alpha;
      }

      // C-layout -> A-layout transpose of P through wave-private LDS
#pragma unroll
      for (int nt = 0; nt < 4; ++nt)
#pragma unroll
        for (int r = 0; r < 8; ++r)
          Ps[wave][r + (grp << 3)][nt * 16 + laneN] = (__bf16)s[nt][r];
      asm volatile("s_wait_dscnt 0x0" ::: "memory");  // same-wave LDS RAW

      // O += P V
#pragma unroll
      for (int ks = 0; ks < 2; ++ks) {
        V16 pa;
        pa.h[0] = *(const v8bf*)&Ps[wave][laneN][ks * 32 + grp * 8];
        pa.h[1] = *(const v8bf*)&Ps[wave][laneN][ks * 32 + 16 + grp * 8];
#pragma unroll
        for (int nt = 0; nt < 4; ++nt) {
          V16 vf;
          const bf16* vr = &Vs[nt * 16 + laneN][ks * 32 + grp * 8];
          vf.h[0] = *(const v8bf*)(vr + 0);
          vf.h[1] = *(const v8bf*)(vr + 16);
          o[nt] = wmma_bf16(pa.v, vf.v, o[nt]);
        }
      }
    }
  }

  // normalize and store Y as [B,T,N] bf16 for the projection GEMM
#pragma unroll
  for (int r = 0; r < 8; ++r) {
    const float inv = 1.0f / l_s[r];
    const int qg = q0w + r + (grp << 3);
#pragma unroll
    for (int nt = 0; nt < 4; ++nt) {
      const int col = h * 64 + nt * 16 + laneN;
      Yb[((size_t)b * 2048 + qg) * 1024 + col] = (__bf16)(o[nt][r] * inv);
    }
  }
}

// ---------------------------------------------------------------------------
// Kernel 3: out = Y @ W_proj + b_proj. Both operands bf16; A and B tiles
// both staged with async DMA, double-buffered.
// ---------------------------------------------------------------------------
__global__ __launch_bounds__(256) void proj_gemm_kernel(
    const bf16* __restrict__ Y, const bf16* __restrict__ Wt,
    const float* __restrict__ bias, float* __restrict__ Out)
{
  constexpr int LDT = 40;
  __shared__ alignas(16) bf16 As[2][128][LDT];
  __shared__ alignas(16) bf16 Bs[2][128][LDT];

  const int tid = threadIdx.x;
  const int wave = tid >> 5, lane = tid & 31;
  const int laneN = lane & 15, grp = lane >> 4;
  const int m0 = blockIdx.y * 128;
  const int n0 = blockIdx.x * 128;
  const int waveM = wave & 3;
  const int waveN = wave >> 2;

  const v8f vzero = {0.f, 0.f, 0.f, 0.f, 0.f, 0.f, 0.f, 0.f};
  v8f acc[2][4];
#pragma unroll
  for (int i = 0; i < 2; ++i)
#pragma unroll
    for (int j = 0; j < 4; ++j) acc[i][j] = vzero;

  const int arow = tid >> 1, acol = (tid & 1) << 4;
  const int bn = tid >> 1,   bk = (tid & 1) << 4;
  const bf16* aptr = Y + (size_t)(m0 + arow) * 1024 + acol;
  const bf16* bptr = Wt + (size_t)(n0 + bn) * 1024 + bk;

  // Prologue: DMA tile 0.
  async_copy_b128(lds_off(&As[0][arow][acol]),     aptr);
  async_copy_b128(lds_off(&As[0][arow][acol + 8]), aptr + 8);
  async_copy_b128(lds_off(&Bs[0][bn][bk]),         bptr);
  async_copy_b128(lds_off(&Bs[0][bn][bk + 8]),     bptr + 8);
  wait_async0();
  __syncthreads();

  int cur = 0;
  for (int k0 = 0; k0 < 1024; k0 += 32) {
    const int nxt = cur ^ 1;
    const bool has_next = (k0 + 32) < 1024;
    if (has_next) {
      async_copy_b128(lds_off(&As[nxt][arow][acol]),     aptr + k0 + 32);
      async_copy_b128(lds_off(&As[nxt][arow][acol + 8]), aptr + k0 + 40);
      async_copy_b128(lds_off(&Bs[nxt][bn][bk]),         bptr + k0 + 32);
      async_copy_b128(lds_off(&Bs[nxt][bn][bk + 8]),     bptr + k0 + 40);
    }

    V16 af[2];
#pragma unroll
    for (int mt = 0; mt < 2; ++mt) {
      const int r = waveM * 32 + mt * 16 + laneN;
      af[mt].h[0] = *(const v8bf*)&As[cur][r][grp * 8];
      af[mt].h[1] = *(const v8bf*)&As[cur][r][16 + grp * 8];
    }
#pragma unroll
    for (int nt = 0; nt < 4; ++nt) {
      V16 bfr;
      const int c = waveN * 64 + nt * 16 + laneN;
      bfr.h[0] = *(const v8bf*)&Bs[cur][c][grp * 8];
      bfr.h[1] = *(const v8bf*)&Bs[cur][c][16 + grp * 8];
#pragma unroll
      for (int mt = 0; mt < 2; ++mt)
        acc[mt][nt] = wmma_bf16(af[mt].v, bfr.v, acc[mt][nt]);
    }

    wait_async0();
    __syncthreads();
    cur = nxt;
  }

#pragma unroll
  for (int nt = 0; nt < 4; ++nt) {
    const int gn = n0 + waveN * 64 + nt * 16 + laneN;
    const float bv = bias[gn];
#pragma unroll
    for (int mt = 0; mt < 2; ++mt) {
#pragma unroll
      for (int r = 0; r < 8; ++r) {
        const int gm = m0 + waveM * 32 + mt * 16 + r + (grp << 3);
        Out[(size_t)gm * 1024 + gn] = acc[mt][nt][r] + bv;
      }
    }
  }
}

// ---------------------------------------------------------------------------
extern "C" void kernel_launch(void* const* d_in, const int* in_sizes, int n_in,
                              void* d_out, int out_size, void* d_ws, size_t ws_size,
                              hipStream_t stream) {
  (void)in_sizes; (void)n_in; (void)out_size; (void)ws_size;
  const float* x      = (const float*)d_in[0];
  const float* W_attn = (const float*)d_in[1];
  const float* b_attn = (const float*)d_in[2];
  const float* W_proj = (const float*)d_in[3];
  const float* b_proj = (const float*)d_in[4];
  float* out = (float*)d_out;

  char* ws = (char*)d_ws;
  const size_t SZ = (size_t)4 * 16 * 2048 * 64 * sizeof(bf16);  // 16 MiB
  bf16* Qb  = (bf16*)(ws);
  bf16* Kb  = (bf16*)(ws + SZ);
  bf16* Vb  = (bf16*)(ws + 2 * SZ);
  bf16* Yb  = (bf16*)(ws + 3 * SZ);
  bf16* WtA = (bf16*)(ws + 4 * SZ);                       // 3072x1024 bf16, 6 MiB
  bf16* WtP = (bf16*)(ws + 4 * SZ + (size_t)6 * 1024 * 1024);  // 1024x1024 bf16

  // Weight transpose+convert (one-time, amortized over the GEMMs).
  wt_convert_kernel<<<dim3(96, 32), 256, 0, stream>>>(W_attn, WtA, 1024, 3072);
  wt_convert_kernel<<<dim3(32, 32), 256, 0, stream>>>(W_proj, WtP, 1024, 1024);
  // QKV: M=8192, N=3072 -> grid (24, 64)
  qkv_gemm_kernel<<<dim3(24, 64), 256, 0, stream>>>(x, WtA, b_attn, Qb, Kb, Vb);
  // Attention: grid (B*H, T/128)
  flash_attn_kernel<<<dim3(64, 16), 256, 0, stream>>>(Qb, Kb, Vb, Yb);
  // Projection: M=8192, N=1024 -> grid (8, 64)
  proj_gemm_kernel<<<dim3(8, 64), 256, 0, stream>>>(Yb, WtP, b_proj, out);
}
